// GCN_26834955665843
// MI455X (gfx1250) — matrix operand order repaired and checked by reading, not compiled, not run on hardware
//
#include <hip/hip_runtime.h>
#include <stdint.h>

// ---------------------------------------------------------------------------
// GatedGraphConv x2 on MI455X (gfx1250).
//  - All 5 GEMM shapes on v_wmma_f32_16x16x32_bf16 (bf16 in, f32 accum).
//  - One wave = one 16-row panel, A fragments register-resident across all
//    column tiles (A read exactly once per GEMM from HBM/L2).
//  - Packed-B weight panel staged into LDS per block via the Tensor Data
//    Mover (tensor_load_to_lds + s_wait_tensorcnt); B fragments consumed via
//    ds_load_b128 on the DS pipe, co-issuing with WMMA.
//  - Edge scatter-add as f32 L2 atomics (m fits in the 192 MB L2).
// ---------------------------------------------------------------------------

#define D_IN   128
#define H_DIM  128
#define D_OUT3 384
#define D_OUT  32
#define KT     4            // K = 128 -> 4 WMMA K-steps
#define MAX_NT 12           // max column tiles per sub-GEMM (192 cols)
#define PANEL_U16 (KT * MAX_NT * 512)   // 24576 u16 = 48 KB LDS

typedef __attribute__((ext_vector_type(16))) __bf16 v16bf;
typedef __attribute__((ext_vector_type(8)))  float  v8f;
typedef __attribute__((ext_vector_type(4)))  uint32_t v4u;
typedef __attribute__((ext_vector_type(4)))  int      v4i;
typedef __attribute__((ext_vector_type(8)))  int      v8i;
static_assert(sizeof(v16bf) == 32, "v16bf must be 32 bytes");

union FragBF { uint4 u[2]; v16bf v; };

#if defined(__has_builtin)
#  if __has_builtin(__builtin_amdgcn_tensor_load_to_lds)
#    define HAVE_TDM 1
#  endif
#  if __has_builtin(__builtin_amdgcn_s_wait_tensorcnt)
#    define HAVE_TENSORCNT 1
#  endif
#endif

__device__ __forceinline__ uint16_t f2bf(float f) {
  union { float f; uint32_t u; } v; v.f = f;
  uint32_t u = v.u;
  return (uint16_t)((u + 0x7FFFu + ((u >> 16) & 1u)) >> 16);  // RNE
}

__device__ __forceinline__ void wait_tensorcnt0() {
#if defined(HAVE_TENSORCNT)
  __builtin_amdgcn_s_wait_tensorcnt(0);
#else
  asm volatile("s_wait_tensorcnt 0x0" ::: "memory");
#endif
}

// ---------------------------------------------------------------------------
// Pack fp32 weight into bf16 WMMA-B fragment-native layout.
// Logical B is [128 x (ncolOff + nt*16 + ...)]. Per (ktile, ntile): 512 bf16
// so lane L reads its 16 B-fragment values as one contiguous 32B run:
//   k = kt*32 + (L>=16 ? 16 : 0) + v,  n = ncolOff + nt*16 + (L & 15)
// transpose=1: src is [cols x 128] row-major (GRU wih/whh used as W^T).
// ---------------------------------------------------------------------------
__global__ void pack_b_kernel(const float* __restrict__ src, uint16_t* __restrict__ dst,
                              int nt, int ncolOff, int srcLd, int transpose) {
  const int total = KT * nt * 512;
  for (int idx = blockIdx.x * blockDim.x + threadIdx.x; idx < total;
       idx += gridDim.x * blockDim.x) {
    const int v    = idx & 15;
    const int lane = (idx >> 4) & 31;
    const int tile = idx >> 9;
    const int tn   = tile % nt;
    const int tk   = tile / nt;
    const int k = (tk << 5) + ((lane & 16) ? 16 : 0) + v;
    const int n = ncolOff + (tn << 4) + (lane & 15);
    const float val = transpose ? src[(size_t)n * 128 + k]
                                : src[(size_t)k * srcLd + n];
    dst[idx] = f2bf(val);
  }
}

__global__ void cast_bf16_kernel(const float* __restrict__ src, uint16_t* __restrict__ dst,
                                 int validElems, int totalElems) {
  for (int i = blockIdx.x * blockDim.x + threadIdx.x; i < totalElems;
       i += gridDim.x * blockDim.x)
    dst[i] = (i < validElems) ? f2bf(src[i]) : (uint16_t)0;
}

__global__ void zero_u16_kernel(uint16_t* __restrict__ p, int n) {
  for (int i = blockIdx.x * blockDim.x + threadIdx.x; i < n; i += gridDim.x * blockDim.x)
    p[i] = 0;
}

// ---------------------------------------------------------------------------
// Panel GEMM: C[M x cols] = A[M x 128] * B  (+bias, +relu).
// Block = 8 waves; wave w owns row-tile tm = blockIdx*8 + w (M padded to 128).
// Stage: TDM copies the whole packed-B panel (KT*nt*1024 bytes) into LDS;
// meanwhile each wave pulls its 4 A fragments (2x b128 per K-step) into
// registers. Then loop tn: 8x ds_load_b128 + 4x chained WMMA per column tile.
// ---------------------------------------------------------------------------
__global__ void __launch_bounds__(256)
wmma_gemm_kernel(const uint16_t* __restrict__ A, const uint16_t* __restrict__ Bp,
                 int Mvalid, int nt, int ldOut, int ncolOff,
                 const float* __restrict__ bias, int relu,
                 float* __restrict__ outF, uint16_t* __restrict__ outB, int ldOutB) {
  __shared__ __align__(16) uint16_t sB[PANEL_U16];
  const int lane = threadIdx.x & 31;
  const int wave = threadIdx.x >> 5;

  // ---- issue TDM: global packed-B panel -> LDS (one descriptor per block) --
#if defined(HAVE_TDM)
  if (threadIdx.x == 0) {
    const uint32_t nUnits = (uint32_t)(KT * nt * 512 * 2 / 8);  // 8-byte units
    const uint64_t ga  = (uint64_t)Bp;
    const uint32_t lds = (uint32_t)(size_t)(void*)sB;           // LDS byte offset
    // D# group0: count=1 | lds_addr | global_addr | type=2 (bits 127:126)
    v4u g0 = { 1u, lds, (uint32_t)ga, (uint32_t)(ga >> 32) | 0x80000000u };
    // D# group1: data_size=8B; 1-row 2D tile of nUnits elements
    v8i g1 = { (int)(3u << 16),                 // wg_mask=0, data_size=3 (8B)
               (int)((nUnits & 0xFFFFu) << 16), // tensor_dim0[15:0]
               (int)((1u << 16) | (nUnits >> 16)), // tensor_dim1=1 | dim0 hi
               (int)(nUnits << 16),             // tile_dim0 (dim1 hi = 0)
               1,                               // tile_dim1=1, tile_dim2=0
               (int)nUnits,                     // tensor_dim0_stride lo
               0, 0 };                          // strides hi / dim1_stride
    v4i gz = { 0, 0, 0, 0 };
#if __clang_major__ >= 23
    v8i gz8 = { 0, 0, 0, 0, 0, 0, 0, 0 };
    __builtin_amdgcn_tensor_load_to_lds(g0, g1, gz, gz, gz8, 0);
#else
    __builtin_amdgcn_tensor_load_to_lds(g0, g1, gz, gz, 0);
#endif
  }
#else
  // Fallback: cooperative copy (ds_store path).
  {
    const int chunks = (KT * nt * 512) >> 3;  // uint4 chunks
    for (int i = threadIdx.x; i < chunks; i += 256)
      *(uint4*)(sB + (i << 3)) = *(const uint4*)(Bp + (i << 3));
  }
#endif

  // ---- overlap: A fragments into registers while the DMA runs --------------
  const int tm   = blockIdx.x * 8 + wave;          // 16-row panel owned by wave
  const int m    = (tm << 4) + (lane & 15);
  const int koff = (lane & 16) ? 16 : 0;
  const uint16_t* __restrict__ arow = A + (size_t)m * 128 + koff;
  FragBF a[KT];
#pragma unroll
  for (int tk = 0; tk < KT; ++tk) {
    a[tk].u[0] = *(const uint4*)(arow + (tk << 5));
    a[tk].u[1] = *(const uint4*)(arow + (tk << 5) + 8);
  }

#if defined(HAVE_TDM)
  if (threadIdx.x == 0) wait_tensorcnt0();
#endif
  __syncthreads();

  const int rbase = (tm << 4) + ((lane & 16) ? 8 : 0);

  for (int tn = 0; tn < nt; ++tn) {
    v8f c = {0.f, 0.f, 0.f, 0.f, 0.f, 0.f, 0.f, 0.f};
#pragma unroll
    for (int tk = 0; tk < KT; ++tk) {
      FragBF b;
      const uint16_t* bp = sB + (((tk * nt + tn) << 9) + (lane << 4));
      b.u[0] = *(const uint4*)(bp);       // ds_load_b128
      b.u[1] = *(const uint4*)(bp + 8);   // ds_load_b128
      c = __builtin_amdgcn_wmma_f32_16x16x32_bf16(
          false, a[tk].v, false, b.v, (short)0, c, false, false);
    }
    const int col = ncolOff + (tn << 4) + (lane & 15);
    const float bv = bias ? bias[col] : 0.f;
#pragma unroll
    for (int r = 0; r < 8; ++r) {
      const int row = rbase + r;
      if (row < Mvalid) {
        float val = c[r] + bv;
        if (relu) val = fmaxf(val, 0.f);
        outF[(size_t)row * ldOut + col] = val;
        if (outB) outB[(size_t)row * ldOutB + col] = f2bf(val);
      }
    }
  }
}

// ---------------------------------------------------------------------------
// Edge scatter-add: one wave per edge, 4 floats/lane; f32 atomics hit L2.
// ---------------------------------------------------------------------------
__global__ void scatter_edges_kernel(const int* __restrict__ srcIdx,
                                     const int* __restrict__ dstIdx,
                                     const float* __restrict__ msg,
                                     float* __restrict__ agg, int E) {
  const int total = E * 32;
  for (int t = blockIdx.x * blockDim.x + threadIdx.x; t < total;
       t += gridDim.x * blockDim.x) {
    const int e    = t >> 5;
    const int lane = t & 31;
    const int s = srcIdx[e];
    const int d = dstIdx[e];
    const float4 vm = *(const float4*)(msg + (size_t)s * H_DIM + lane * 4);
    float* ap = agg + (size_t)d * H_DIM + lane * 4;
    unsafeAtomicAdd(ap + 0, vm.x);
    unsafeAtomicAdd(ap + 1, vm.y);
    unsafeAtomicAdd(ap + 2, vm.z);
    unsafeAtomicAdd(ap + 3, vm.w);
  }
}

// ---------------------------------------------------------------------------
// GRU update: fuses biases, gates, blend; emits fp32 h + bf16 h (zero pads).
// ---------------------------------------------------------------------------
__global__ void gru_update_kernel(const float* __restrict__ gi, const float* __restrict__ gh,
                                  const float* __restrict__ bih, const float* __restrict__ bhh,
                                  float* __restrict__ h, uint16_t* __restrict__ hb,
                                  int nValidRows, int nPadRows) {
  const int i = blockIdx.x * blockDim.x + threadIdx.x;
  if (i >= nPadRows * H_DIM) return;
  const int row = i >> 7;
  const int col = i & 127;
  if (row < nValidRows) {
    const size_t b3 = (size_t)row * D_OUT3;
    const float ir = gi[b3 + col]       + bih[col];
    const float iz = gi[b3 + 128 + col] + bih[128 + col];
    const float in = gi[b3 + 256 + col] + bih[256 + col];
    const float hr = gh[b3 + col]       + bhh[col];
    const float hz = gh[b3 + 128 + col] + bhh[128 + col];
    const float hn = gh[b3 + 256 + col] + bhh[256 + col];
    const float r = 1.f / (1.f + __expf(-(ir + hr)));
    const float z = 1.f / (1.f + __expf(-(iz + hz)));
    const float n = tanhf(in + r * hn);
    const float hv = h[i];
    const float out = (1.f - z) * n + z * hv;
    h[i]  = out;
    hb[i] = f2bf(out);
  } else {
    hb[i] = 0;
  }
}

// ---------------------------------------------------------------------------

static inline int ceil_div(int a, int b) { return (a + b - 1) / b; }

extern "C" void kernel_launch(void* const* d_in, const int* in_sizes, int n_in,
                              void* d_out, int out_size, void* d_ws, size_t ws_size,
                              hipStream_t stream) {
  const float* x     = (const float*)d_in[0];
  const int*   ei    = (const int*)  d_in[1];
  const float* W_in  = (const float*)d_in[2];
  const float* b_in  = (const float*)d_in[3];
  const float* Wmsg[2] = { (const float*)d_in[4],  (const float*)d_in[9]  };
  const float* wih[2]  = { (const float*)d_in[5],  (const float*)d_in[10] };
  const float* whh[2]  = { (const float*)d_in[6],  (const float*)d_in[11] };
  const float* bih[2]  = { (const float*)d_in[7],  (const float*)d_in[12] };
  const float* bhh[2]  = { (const float*)d_in[8],  (const float*)d_in[13] };
  const float* W_out = (const float*)d_in[14];
  const float* b_out = (const float*)d_in[15];
  float* out = (float*)d_out;

  const int N   = in_sizes[0] / D_IN;        // 50000
  const int E   = in_sizes[1] / 2;           // 800000
  const int Np  = (N + 127) & ~127;          // rows padded to 128 (50048)
  const int nBlk = Np / 128;                 // GEMM grid (8 waves/block)

  // ---- workspace carve-out -------------------------------------------------
  char* ws = (char*)d_ws;
  size_t off = 0;
  auto carve = [&](size_t bytes) -> void* {
    void* p = ws + off;
    off = (off + bytes + 255) & ~(size_t)255;
    return p;
  };
  uint16_t* xb   = (uint16_t*)carve((size_t)Np * H_DIM * 2);
  uint16_t* hb   = (uint16_t*)carve((size_t)Np * H_DIM * 2);
  uint16_t* aggb = (uint16_t*)carve((size_t)Np * H_DIM * 2);
  float*    hf   = (float*)   carve((size_t)N  * H_DIM * 4);
  float*    mf   = (float*)   carve((size_t)N  * H_DIM * 4);
  float*    aggf = (float*)   carve((size_t)Np * H_DIM * 4);
  float*    gif  = (float*)   carve((size_t)N  * D_OUT3 * 4);
  float*    ghf  = (float*)   carve((size_t)N  * D_OUT3 * 4);
  uint16_t* WinP = (uint16_t*)carve((size_t)KT * 8 * 512 * 2);
  uint16_t* WmsgP[2] = { (uint16_t*)carve((size_t)KT * 8 * 512 * 2),
                         (uint16_t*)carve((size_t)KT * 8 * 512 * 2) };
  uint16_t* wihP[2][2], *whhP[2][2];
  for (int l = 0; l < 2; ++l)
    for (int hhalf = 0; hhalf < 2; ++hhalf) {
      wihP[l][hhalf] = (uint16_t*)carve((size_t)KT * MAX_NT * 512 * 2);
      whhP[l][hhalf] = (uint16_t*)carve((size_t)KT * MAX_NT * 512 * 2);
    }
  uint16_t* WoutP = (uint16_t*)carve((size_t)KT * 2 * 512 * 2);
  (void)ws_size; (void)n_in; (void)out_size;

  // ---- 1. pack weights into bf16 fragment layout ---------------------------
  pack_b_kernel<<<ceil_div(KT * 8 * 512, 256), 256, 0, stream>>>(W_in, WinP, 8, 0, 128, 0);
  for (int l = 0; l < 2; ++l) {
    pack_b_kernel<<<ceil_div(KT * 8 * 512, 256), 256, 0, stream>>>(Wmsg[l], WmsgP[l], 8, 0, 128, 0);
    for (int hhalf = 0; hhalf < 2; ++hhalf) {
      pack_b_kernel<<<ceil_div(KT * MAX_NT * 512, 256), 256, 0, stream>>>(
          wih[l], wihP[l][hhalf], MAX_NT, hhalf * 192, 128, 1);
      pack_b_kernel<<<ceil_div(KT * MAX_NT * 512, 256), 256, 0, stream>>>(
          whh[l], whhP[l][hhalf], MAX_NT, hhalf * 192, 128, 1);
    }
  }
  pack_b_kernel<<<ceil_div(KT * 2 * 512, 256), 256, 0, stream>>>(W_out, WoutP, 2, 0, 32, 0);

  // ---- 2. activations to bf16; zero hb pad rows ----------------------------
  const int elemsPad = Np * H_DIM;
  cast_bf16_kernel<<<ceil_div(elemsPad, 256), 256, 0, stream>>>(x, xb, N * H_DIM, elemsPad);
  if (Np > N)
    zero_u16_kernel<<<ceil_div((Np - N) * H_DIM, 256), 256, 0, stream>>>(
        hb + (size_t)N * H_DIM, (Np - N) * H_DIM);

  // ---- 3. h = relu(x @ W_in + b_in) ---------------------------------------
  wmma_gemm_kernel<<<nBlk, 256, 0, stream>>>(
      xb, WinP, N, 8, 128, 0, b_in, 1, hf, hb, 128);

  // ---- 4. two gated layers -------------------------------------------------
  for (int l = 0; l < 2; ++l) {
    // m = h @ W_msg
    wmma_gemm_kernel<<<nBlk, 256, 0, stream>>>(
        hb, WmsgP[l], N, 8, 128, 0, nullptr, 0, mf, nullptr, 0);
    // agg = segment_sum(m[src], dst)
    hipMemsetAsync(aggf, 0, (size_t)Np * H_DIM * 4, stream);
    scatter_edges_kernel<<<ceil_div(E * 32, 256), 256, 0, stream>>>(
        ei, ei + E, mf, aggf, E);
    cast_bf16_kernel<<<ceil_div(elemsPad, 256), 256, 0, stream>>>(
        aggf, aggb, elemsPad, elemsPad);
    // gi = agg @ wih^T ; gh = h @ whh^T  (two 192-col halves each)
    for (int hhalf = 0; hhalf < 2; ++hhalf) {
      wmma_gemm_kernel<<<nBlk, 256, 0, stream>>>(
          aggb, wihP[l][hhalf], N, MAX_NT, D_OUT3, hhalf * 192, nullptr, 0, gif, nullptr, 0);
      wmma_gemm_kernel<<<nBlk, 256, 0, stream>>>(
          hb, whhP[l][hhalf], N, MAX_NT, D_OUT3, hhalf * 192, nullptr, 0, ghf, nullptr, 0);
    }
    // GRU update -> hf (fp32) + hb (bf16)
    gru_update_kernel<<<ceil_div(elemsPad, 256), 256, 0, stream>>>(
        gif, ghf, bih[l], bhh[l], hf, hb, N, Np);
  }

  // ---- 5. out = h @ W_out + b_out -----------------------------------------
  wmma_gemm_kernel<<<nBlk, 256, 0, stream>>>(
      hb, WoutP, N, 2, D_OUT, 0, b_out, 0, out, nullptr, 0);
}
